// WeightedKspaceMSELossNonCart_3917010174176
// MI455X (gfx1250) — matrix operand order repaired and checked by reading, not compiled, not run on hardware
//
#include <hip/hip_runtime.h>
#include <hip/hip_bf16.h>

// ---------------------------------------------------------------------------
// WeightedKspaceMSELossNonCart on gfx1250 (MI455X)
//
// k[c,m] = (1/300) * sum_{h,w} sim[c,h,w] * ez[m,w] * ey[m,h]
// Stage 1 (1.18e10 flops): complex GEMM ez(M x W) * simT(W x H) via
//   v_wmma_f32_16x16x32_f16 (4 real accumulators).
// Stage 2: ey[m,h] fused into the 16x16 D tile in-register.
//
// CDNA5 data movement: the sim h-tile slab (16 x 320 f16, shared by all 4
// waves of a block) is DMA'd into LDS by the Tensor Data Mover
// (tensor_load_to_lds, double-buffered, s_wait_tensorcnt + barrier), so B
// fragments come from ds_load_b128 instead of 4x-redundant global loads.
// ---------------------------------------------------------------------------

typedef __attribute__((ext_vector_type(16))) _Float16 v16h;
typedef __attribute__((ext_vector_type(8)))  _Float16 v8h;
typedef __attribute__((ext_vector_type(8)))  float    v8f;
typedef __attribute__((ext_vector_type(4)))  unsigned int u32x4;
typedef __attribute__((ext_vector_type(8)))  int i32x8;
typedef __attribute__((ext_vector_type(4)))  int i32x4;

#define HDIM 300
#define WDIM 300
#define MLEN 4096
#define CNC  4
#define WP   320   // padded W: 10 k-steps of 32
#define HP   304   // padded H: 19 tiles of 16
#define TWO_PI 6.28318530717958647692f

#if __has_include(<hip/amd_detail/amd_gfx1250_TDM.h>)
#define TDM_6ARG 1
#else
#define TDM_6ARG 0
#endif

static __device__ __forceinline__ v16h ld16(const _Float16* p0, const _Float16* p1) {
  v8h a = *(const v8h*)p0;   // 16B contiguous run
  v8h b = *(const v8h*)p1;   // 16B contiguous run
  return __builtin_shufflevector(a, b, 0,1,2,3,4,5,6,7,8,9,10,11,12,13,14,15);
}

static __device__ __forceinline__ v8f wmma16(v16h a, v16h b, v8f c) {
  // D = A(16x32 f16) * B(32x16 f16) + C(16x16 f32)
  return __builtin_amdgcn_wmma_f32_16x16x32_f16(false, a, false, b, (short)0, c,
                                                false, false);
}

// Issue one 2D TDM descriptor: rows x cols f16 tile, global (row-major,
// stride = cols elements) -> LDS at byte offset lds_addr.
static __device__ __forceinline__ void tdm_load_2d(unsigned lds_addr,
                                                   const _Float16* gsrc,
                                                   int rows, int cols) {
  unsigned long long ga = (unsigned long long)(uintptr_t)gsrc;
  u32x4 g0;
  g0.x = 1u;                                              // count=1 (valid user D#)
  g0.y = lds_addr;                                        // lds_addr (bytes)
  g0.z = (unsigned)(ga & 0xFFFFFFFFu);                    // global_addr[31:0]
  g0.w = ((unsigned)(ga >> 32) & 0x01FFFFFFu) | 0x80000000u; // [56:32] | type=2
  i32x8 g1;
  g1[0] = 1 << 16;            // workgroup_mask=0, data_size=1 (2 bytes)
  g1[1] = cols << 16;         // atomic_barrier=0 | tensor_dim0[15:0]
  g1[2] = rows << 16;         // tensor_dim0[31:16]=0 | tensor_dim1[15:0]
  g1[3] = cols << 16;         // tensor_dim1[31:16]=0 | tile_dim0
  g1[4] = rows;               // tile_dim1 | tile_dim2=0
  g1[5] = cols;               // tensor_dim0_stride[31:0]
  g1[6] = 0;                  // stride0 hi | tensor_dim1_stride lo (unused, 2D)
  g1[7] = 0;
  i32x4 z4 = {0, 0, 0, 0};
#if TDM_6ARG
  i32x8 z8 = {0, 0, 0, 0, 0, 0, 0, 0};
  __builtin_amdgcn_tensor_load_to_lds(g0, g1, z4, z4, z8, 0);
#else
  __builtin_amdgcn_tensor_load_to_lds(g0, g1, z4, z4, 0);
#endif
}

static __device__ __forceinline__ unsigned lds_byte_off(const void* p) {
  // generic LDS pointer: low 32 bits hold the LDS byte address
  return (unsigned)(uintptr_t)p;
}

// ---------------- precompute kernels (build L2-resident tables) ----------------

__global__ void zero_kbuf(float* kbuf, int n) {
  int i = blockIdx.x * blockDim.x + threadIdx.x;
  if (i < n) kbuf[i] = 0.0f;
}

__global__ void build_ez(const float* __restrict__ omega1,
                         _Float16* __restrict__ ezr, _Float16* __restrict__ ezi) {
  int idx = blockIdx.x * blockDim.x + threadIdx.x;
  if (idx >= MLEN * WP) return;
  int m = idx / WP, w = idx % WP;
  float re = 0.0f, im = 0.0f;
  if (w < WDIM) {
    float t = TWO_PI * omega1[m] * ((float)w - 150.0f);
    float s, c; __sincosf(t, &s, &c);
    re = c; im = -s;                       // exp(-i t)
  }
  ezr[idx] = (_Float16)re;
  ezi[idx] = (_Float16)im;
}

__global__ void build_ey(const float* __restrict__ omega0, float2* __restrict__ ey) {
  int idx = blockIdx.x * blockDim.x + threadIdx.x;
  if (idx >= MLEN * HP) return;
  int m = idx / HP, h = idx % HP;
  float2 v = make_float2(0.0f, 0.0f);
  if (h < HDIM) {
    float t = TWO_PI * omega0[m] * ((float)h - 150.0f);
    float s, c; __sincosf(t, &s, &c);
    v = make_float2(c, -s);                // exp(-i t)
  }
  ey[idx] = v;
}

__global__ void build_sim(const float* __restrict__ img,     // (300,300,1)
                          const float* __restrict__ csm,     // (4,300,300,2)
                          _Float16* __restrict__ smr, _Float16* __restrict__ smi) {
  int idx = blockIdx.x * blockDim.x + threadIdx.x;
  if (idx >= CNC * HP * WP) return;
  int w = idx % WP;
  int t = idx / WP;
  int h = t % HP;
  int c = t / HP;
  float re = 0.0f, im = 0.0f;
  if (h < HDIM && w < WDIM) {
    float iv = img[h * WDIM + w];
    size_t o = ((size_t)c * HDIM * WDIM + (size_t)h * WDIM + w) * 2;
    re = csm[o] * iv;
    im = csm[o + 1] * iv;
  }
  smr[idx] = (_Float16)re;
  smi[idx] = (_Float16)im;
}

// ---------------- main kernel: TDM -> LDS -> WMMA complex GEMM ----------------
// block = 128 threads = 4 waves; wave owns a 16-m tile.
// grid = (M/64, C, 2); gridDim.z splits the 19 h-tiles (exactly 2 atomic adds
// per kbuf entry -> commutative -> bitwise deterministic).

__global__ __launch_bounds__(128) void nufft_kspace_wmma(
    const _Float16* __restrict__ ezr, const _Float16* __restrict__ ezi,
    const _Float16* __restrict__ smr, const _Float16* __restrict__ smi,
    const float2*  __restrict__ ey,
    float* __restrict__ kbuf /* (C, M, 2) f32 */) {
  __shared__ _Float16 lds_sr[2][16 * WP];   // 2 x 10KB
  __shared__ _Float16 lds_si[2][16 * WP];   // 2 x 10KB

  const int lane = threadIdx.x & 31;
  const int wave = threadIdx.x >> 5;
  const int hs   = lane >> 4;      // lane half: selects K sub-range of fragments
  const int l15  = lane & 15;
  const int c    = blockIdx.y;
  const int m_base = blockIdx.x * 64 + wave * 16;

  float kre[8] = {0,0,0,0,0,0,0,0};
  float kim[8] = {0,0,0,0,0,0,0,0};

  // A operand (ez): lane row = l15; runs at K = {0..7,16..23} (lanes 0-15)
  //                                        K = {8..15,24..31} (lanes 16-31)
  const _Float16* arow_r = ezr + (size_t)(m_base + l15) * WP;
  const _Float16* arow_i = ezi + (size_t)(m_base + l15) * WP;
  const int aoff = hs ? 8 : 0;

  const int ht0 = blockIdx.z;
  // Prologue: TDM the first sim h-slab into buffer 0 (wave 0 only; TDM
  // ignores EXEC, so it must be issued by exactly one wave).
  if (wave == 0) {
    const size_t slab = ((size_t)c * HP + (size_t)ht0 * 16) * WP;
    tdm_load_2d(lds_byte_off(&lds_sr[0][0]), smr + slab, 16, WP);
    tdm_load_2d(lds_byte_off(&lds_si[0][0]), smi + slab, 16, WP);
  }

  int buf = 0;
  for (int ht = ht0; ht < 19; ht += 2) {
    if (wave == 0) __builtin_amdgcn_s_wait_tensorcnt(0);  // slab `ht` landed
    __syncthreads();                                      // release all waves
    // Overlap: DMA the next slab into the other buffer while computing.
    if (wave == 0 && ht + 2 < 19) {
      const size_t slab = ((size_t)c * HP + (size_t)(ht + 2) * 16) * WP;
      tdm_load_2d(lds_byte_off(&lds_sr[buf ^ 1][0]), smr + slab, 16, WP);
      tdm_load_2d(lds_byte_off(&lds_si[buf ^ 1][0]), smi + slab, 16, WP);
    }

    const int h_base = ht * 16;
    v8f P1 = {}, P2 = {}, P3 = {}, P4 = {};
    // B operand (sim slab in LDS): lane col = l15 (row h); K = {0..7,16..23} / {8..15,24..31}
    const int bidx = l15 * WP + 8 * hs;
#pragma unroll 2
    for (int wt = 0; wt < 10; ++wt) {
      const int wb = wt * 32;
      v16h ar = ld16(arow_r + wb + aoff, arow_r + wb + aoff + 16);
      v16h ai = ld16(arow_i + wb + aoff, arow_i + wb + aoff + 16);
      v16h br = ld16(&lds_sr[buf][bidx + wb], &lds_sr[buf][bidx + wb + 16]);
      v16h bi = ld16(&lds_si[buf][bidx + wb], &lds_si[buf][bidx + wb + 16]);
      P1 = wmma16(ar, br, P1);   // ReA*ReB
      P2 = wmma16(ai, bi, P2);   // ImA*ImB
      P3 = wmma16(ar, bi, P3);   // ReA*ImB
      P4 = wmma16(ai, br, P4);   // ImA*ReB
    }
    // Fuse ey[m,h] into the 16x16 tile.  D layout: VGPR r, lanes 0-15 -> row r,
    // col = lane; lanes 16-31 -> row r+8, col = lane-16.
#pragma unroll
    for (int r = 0; r < 8; ++r) {
      float2 e = ey[(size_t)(m_base + r + 8 * hs) * HP + h_base + l15];
      float Re = P1[r] - P2[r];
      float Im = P3[r] + P4[r];
      kre[r] += Re * e.x - Im * e.y;
      kim[r] += Re * e.y + Im * e.x;
    }
    buf ^= 1;
  }

  // Reduce the 16 columns (butterfly within each 16-lane half).
#pragma unroll
  for (int r = 0; r < 8; ++r) {
    for (int off = 1; off < 16; off <<= 1) {
      kre[r] += __shfl_xor(kre[r], off, 32);
      kim[r] += __shfl_xor(kim[r], off, 32);
    }
  }
  if (l15 == 0) {
#pragma unroll
    for (int r = 0; r < 8; ++r) {
      int m = m_base + r + 8 * hs;
      atomicAdd(&kbuf[((size_t)c * MLEN + m) * 2 + 0], kre[r]);
      atomicAdd(&kbuf[((size_t)c * MLEN + m) * 2 + 1], kim[r]);
    }
  }
}

// ---------------- final deterministic reduction: weighted MSE ----------------

__global__ __launch_bounds__(256) void finalize_loss(
    const float* __restrict__ kbuf, const float* __restrict__ omega,
    const float* __restrict__ kdata, float* __restrict__ out) {
  __shared__ float red[256];
  const float scale = 1.0f / 300.0f;     // 1/sqrt(H*W)
  float acc = 0.0f;
  for (int i = threadIdx.x; i < CNC * MLEN; i += 256) {
    int m = i % MLEN;
    float kr = kbuf[i * 2 + 0] * scale;
    float ki = kbuf[i * 2 + 1] * scale;
    float o0 = omega[m] * TWO_PI;
    float o1 = omega[MLEN + m] * TWO_PI;
    float wgt = sqrtf(o0 * o0 + o1 * o1) + 1.0f;
    float dr = wgt * (kr - kdata[i * 2 + 0]);
    float di = wgt * (ki - kdata[i * 2 + 1]);
    acc += dr * dr + di * di;
  }
  red[threadIdx.x] = acc;
  __syncthreads();
  for (int s = 128; s > 0; s >>= 1) {
    if (threadIdx.x < s) red[threadIdx.x] += red[threadIdx.x + s];
    __syncthreads();
  }
  if (threadIdx.x == 0) out[0] = red[0] / (2.0f * CNC * MLEN);
}

// ---------------- launcher ----------------

extern "C" void kernel_launch(void* const* d_in, const int* in_sizes, int n_in,
                              void* d_out, int out_size, void* d_ws, size_t ws_size,
                              hipStream_t stream) {
  const float* img   = (const float*)d_in[0];   // (300,300,1)
  const float* kdata = (const float*)d_in[1];   // (1,4,4096,2)
  const float* omega = (const float*)d_in[2];   // (1,2,4096)
  // d_in[3] = all_dcomp: unused by the reference
  const float* csm   = (const float*)d_in[4];   // (4,300,300,2)
  float* out = (float*)d_out;

  // workspace carve-up (~17 MB, L2-resident)
  _Float16* ezr = (_Float16*)d_ws;
  _Float16* ezi = ezr + (size_t)MLEN * WP;
  _Float16* smr = ezi + (size_t)MLEN * WP;
  _Float16* smi = smr + (size_t)CNC * HP * WP;
  float2*   ey  = (float2*)(smi + (size_t)CNC * HP * WP);   // 8B-aligned offset
  float*    kbuf = (float*)(ey + (size_t)MLEN * HP);        // (C,M,2) f32

  const int nk = CNC * MLEN * 2;
  zero_kbuf<<<(nk + 255) / 256, 256, 0, stream>>>(kbuf, nk);

  const int nez = MLEN * WP;
  build_ez<<<(nez + 255) / 256, 256, 0, stream>>>(omega + MLEN, ezr, ezi);

  const int ney = MLEN * HP;
  build_ey<<<(ney + 255) / 256, 256, 0, stream>>>(omega, ey);

  const int nsim = CNC * HP * WP;
  build_sim<<<(nsim + 255) / 256, 256, 0, stream>>>(img, csm, smr, smi);

  dim3 grid(MLEN / 64, CNC, 2);
  nufft_kspace_wmma<<<grid, 128, 0, stream>>>(ezr, ezi, smr, smi, ey, kbuf);

  finalize_loss<<<1, 256, 0, stream>>>(kbuf, omega, kdata, out);
}
